// FourrierKernelAttention_17334488007446
// MI455X (gfx1250) — matrix operand order repaired and checked by reading, not compiled, not run on hardware
//
#include <hip/hip_runtime.h>
#include <hip/hip_bf16.h>

// Problem constants (from reference): N=2, L=2048, H=8, D=64, DIM=512, P=1
#define N_   2
#define L_   2048
#define H_   8
#define D_   64
#define DIM_ 512

typedef __attribute__((ext_vector_type(16))) __bf16 v16bf;
typedef __attribute__((ext_vector_type(8)))  __bf16 v8bf;
typedef __attribute__((ext_vector_type(8)))  float  v8f;

__device__ __forceinline__ v8f wmma_bf16(v16bf a, v16bf b, v8f c) {
    // D = A(16x32 bf16) * B(32x16 bf16) + C(16x16 f32)
    return __builtin_amdgcn_wmma_f32_16x16x32_bf16(
        false, a, false, b, (short)0, c, false, false);
}

__device__ __forceinline__ float softplus_f(float x) {
    return (x > 20.0f) ? x : __logf(1.0f + __expf(x));
}

// ---------------------------------------------------------------------------
// Pass 0: bulk f32 -> bf16 conversion (8 elements / thread, 16B stores)
// ---------------------------------------------------------------------------
__global__ __launch_bounds__(256)
void fka_cvt_kernel(const float* __restrict__ src, __bf16* __restrict__ dst, int n8)
{
    int i = blockIdx.x * 256 + threadIdx.x;
    if (i < n8) {
        const float4* s = (const float4*)src;
        float4 a = s[i * 2], b = s[i * 2 + 1];
        v8bf o;
        o[0] = (__bf16)a.x; o[1] = (__bf16)a.y; o[2] = (__bf16)a.z; o[3] = (__bf16)a.w;
        o[4] = (__bf16)b.x; o[5] = (__bf16)b.y; o[6] = (__bf16)b.z; o[7] = (__bf16)b.w;
        *(v8bf*)&dst[i * 8] = o;
    }
}

// ---------------------------------------------------------------------------
// Pass 1: projections  Q/K/V = y @ W^T (bf16 WMMA), fused softplus + rotation.
// Inputs already bf16. Writes bf16 QC,QS,KC,KS,V in (N,H,L,D) layout.
// grid = (4096/64, 512/64, 3), block = 128 (4 waves), wave = 16 rows x 64 cols
// ---------------------------------------------------------------------------
__global__ __launch_bounds__(128)
void fka_proj_kernel(const __bf16* __restrict__ ybf,
                     const __bf16* __restrict__ Wqb,
                     const __bf16* __restrict__ Wkb,
                     const __bf16* __restrict__ Wvb,
                     const float* __restrict__ positions,
                     const float* __restrict__ coeffs,   // (H,D)
                     const float* __restrict__ pw,       // (H,D,1)
                     const float* __restrict__ pb,       // (H,D)
                     __bf16* __restrict__ QC, __bf16* __restrict__ QS,
                     __bf16* __restrict__ KC, __bf16* __restrict__ KS,
                     __bf16* __restrict__ Vb)
{
    const int lane  = threadIdx.x & 31;
    const int wave  = threadIdx.x >> 5;
    const int khalf = lane >> 4;
    const int l16   = lane & 15;
    const int z     = blockIdx.z;
    const __bf16* __restrict__ W = (z == 0) ? Wqb : (z == 1) ? Wkb : Wvb;

    const int rbase = blockIdx.x * 64 + wave * 16;
    const int cbase = blockIdx.y * 64;

    v8f acc[4] = {};
    const __bf16* yrow = ybf + (size_t)(rbase + l16) * DIM_;

    for (int ks = 0; ks < DIM_ / 32; ++ks) {
        // A fragment: 16 contiguous bf16 (two 16B vector loads)
        v8bf alo = *(const v8bf*)(yrow + ks * 32 + khalf * 16);
        v8bf ahi = *(const v8bf*)(yrow + ks * 32 + khalf * 16 + 8);
        v16bf a = __builtin_shufflevector(alo, ahi,
            0, 1, 2, 3, 4, 5, 6, 7, 8, 9, 10, 11, 12, 13, 14, 15);
        #pragma unroll
        for (int nt = 0; nt < 4; ++nt) {
            int col = cbase + nt * 16 + l16;
            // B fragment: 16 contiguous bf16 from W row (32B vector load)
            v16bf b = *(const v16bf*)&W[(size_t)col * DIM_ + ks * 32 + khalf * 16];
            acc[nt] = wmma_bf16(a, b, acc[nt]);
        }
    }

    #pragma unroll
    for (int nt = 0; nt < 4; ++nt) {
        int col = cbase + nt * 16 + l16;
        int h = col >> 6, d = col & 63;
        float cf = coeffs[h * D_ + d];
        float w1 = pw[h * D_ + d];
        float b1 = pb[h * D_ + d];
        #pragma unroll
        for (int r = 0; r < 8; ++r) {
            int row = rbase + r + 8 * khalf;
            int n = row >> 11, l = row & (L_ - 1);
            float vacc = acc[nt][r];
            size_t o = ((size_t)(n * H_ + h) * L_ + l) * D_ + d;
            if (z == 2) {
                Vb[o] = (__bf16)vacc;
            } else {
                float sp  = softplus_f(vacc);
                float pos = positions[n * L_ + l];     // P == 1
                float ph  = w1 * pos + b1;
                float cp  = __cosf(ph), sn = __sinf(ph);
                if (z == 0) {
                    float qc = sp * cf;
                    QC[o] = (__bf16)(qc * cp);
                    QS[o] = (__bf16)(qc * sn);
                } else {
                    KC[o] = (__bf16)(sp * cp);
                    KS[o] = (__bf16)(sp * sn);
                }
            }
        }
    }
}

// ---------------------------------------------------------------------------
// Pass 2: causal linear-normalized attention, flash-style.
// grid = (L/128 query tiles, N*H), block = 256 (8 waves), wave = 16 query rows
// ---------------------------------------------------------------------------
__global__ __launch_bounds__(256)
void fka_attn_kernel(const __bf16* __restrict__ QC, const __bf16* __restrict__ QS,
                     const __bf16* __restrict__ KC, const __bf16* __restrict__ KS,
                     const __bf16* __restrict__ Vb, float* __restrict__ out)
{
    __shared__ __align__(32) __bf16 lds_k[64 * 128];   // [key][feat 0..127]  16 KB
    __shared__ __align__(32) __bf16 lds_v[64 * 64];    // [d][key] transposed  8 KB
    __shared__ __align__(32) float  lds_s[8][16 * 64]; // per-wave S scratch  32 KB

    const int t     = threadIdx.x;
    const int lane  = t & 31;
    const int wave  = t >> 5;
    const int khalf = lane >> 4;
    const int l16   = lane & 15;
    const int qt    = blockIdx.x;          // 128-row query tile
    const int bh    = blockIdx.y;          // n*H + h
    const size_t base = (size_t)bh * L_ * D_;

    // Preload Q A-fragments: 16 rows x 128 features ([qc|qs]) in 4 K-steps
    const int wrow0 = qt * 128 + wave * 16;
    const int qrow  = wrow0 + l16;
    v16bf aq[4];
    #pragma unroll
    for (int s = 0; s < 4; ++s) {
        const __bf16* __restrict__ Qsrc = (s < 2) ? QC : QS;
        const __bf16* p = Qsrc + base + (size_t)qrow * D_ + (s & 1) * 32;
        v8bf lo = *(const v8bf*)(p + khalf * 16);
        v8bf hi = *(const v8bf*)(p + khalf * 16 + 8);
        aq[s] = __builtin_shufflevector(lo, hi,
            0, 1, 2, 3, 4, 5, 6, 7, 8, 9, 10, 11, 12, 13, 14, 15);
    }

    v8f accO[4] = {};
    float rowsum = 0.0f;

    const int ktmax = 2 * qt + 1;  // last 64-key tile overlapping this block
    for (int kt = 0; kt <= ktmax; ++kt) {
        // ---- cooperative LDS fill (16B chunks for K, transposed scalar V) ----
        #pragma unroll
        for (int i = 0; i < 4; ++i) {                 // 256 thr x 4 = 1024 chunks
            int c    = t + i * 256;
            int key  = c >> 4;                        // 0..63
            int part = c & 15;                        // feat chunk: part*8
            const __bf16* __restrict__ Ksrc = (part < 8) ? KC : KS;
            const __bf16* src =
                Ksrc + base + (size_t)(kt * 64 + key) * D_ + (part & 7) * 8;
            *(v8bf*)&lds_k[key * 128 + part * 8] = *(const v8bf*)src;
        }
        {
            int d  = t & 63;
            int k0 = (t >> 6) * 16;                   // 4 groups of 16 keys
            for (int kk = 0; kk < 16; ++kk)
                lds_v[d * 64 + k0 + kk] =
                    Vb[base + (size_t)(kt * 64 + k0 + kk) * D_ + d];
        }
        if (kt < ktmax) {  // gfx1250 global prefetch of next key tile
            __builtin_prefetch(&KC[base + (size_t)((kt + 1) * 64) * D_], 0, 1);
            __builtin_prefetch(&Vb[base + (size_t)((kt + 1) * 64) * D_], 0, 1);
        }
        __syncthreads();

        if (kt * 64 <= wrow0 + 15) {   // wave-uniform: tile not fully above rows
            float* smem_s = &lds_s[wave][0];

            // ---- S = [qc|qs] @ [kc|ks]^T : 4 col-tiles x 4 K-steps ----
            #pragma unroll
            for (int nt = 0; nt < 4; ++nt) {
                v8f sc = {};
                #pragma unroll
                for (int s = 0; s < 4; ++s) {
                    const v16bf* bp = (const v16bf*)
                        &lds_k[(nt * 16 + l16) * 128 + s * 32 + khalf * 16];
                    sc = wmma_bf16(aq[s], *bp, sc);
                }
                int kcol = kt * 64 + nt * 16 + l16;
                #pragma unroll
                for (int r = 0; r < 8; ++r) {
                    int row = wrow0 + r + 8 * khalf;
                    float v = (kcol <= row) ? sc[r] : 0.0f;
                    smem_s[(r + 8 * khalf) * 64 + nt * 16 + l16] = v;
                }
            }

            // ---- row sums (lanes 0..15 each own one query row) ----
            if (lane < 16) {
                float a = 0.0f;
                #pragma unroll 8
                for (int j = 0; j < 64; ++j) a += smem_s[lane * 64 + j];
                rowsum += a;
            }

            // ---- accO += S @ V : relayout S (D->A frag) via LDS ----
            #pragma unroll
            for (int s2 = 0; s2 < 2; ++s2) {
                v16bf as;
                #pragma unroll
                for (int i = 0; i < 16; ++i) {
                    int k = s2 * 32 + khalf * 16 + i;
                    as[i] = (__bf16)smem_s[l16 * 64 + k];
                }
                #pragma unroll
                for (int nt = 0; nt < 4; ++nt) {
                    const v16bf* bv = (const v16bf*)
                        &lds_v[(nt * 16 + l16) * 64 + s2 * 32 + khalf * 16];
                    accO[nt] = wmma_bf16(as, *bv, accO[nt]);
                }
            }
        }
        __syncthreads();
    }

    // ---- finalize: divide by row sum, store f32 (N,L,DIM) ----
    int n = bh >> 3, h = bh & 7;
    #pragma unroll
    for (int r = 0; r < 8; ++r) {
        float rs  = __shfl(rowsum, r + 8 * khalf, 32);
        float inv = 1.0f / rs;
        int row = wrow0 + r + 8 * khalf;
        #pragma unroll
        for (int nt = 0; nt < 4; ++nt) {
            int d = nt * 16 + l16;
            out[((size_t)(n * L_ + row)) * DIM_ + h * D_ + d] = accO[nt][r] * inv;
        }
    }
}

extern "C" void kernel_launch(void* const* d_in, const int* in_sizes, int n_in,
                              void* d_out, int out_size, void* d_ws, size_t ws_size,
                              hipStream_t stream) {
    const float* y         = (const float*)d_in[0];
    const float* positions = (const float*)d_in[1];
    const float* Wq        = (const float*)d_in[2];
    const float* Wk        = (const float*)d_in[3];
    const float* Wv        = (const float*)d_in[4];
    const float* coeffs    = (const float*)d_in[5];
    const float* pw        = (const float*)d_in[6];
    const float* pb        = (const float*)d_in[7];

    char* ws = (char*)d_ws;
    const size_t tsz = (size_t)N_ * H_ * L_ * D_ * sizeof(__bf16);  // 4 MB each
    __bf16* QC  = (__bf16*)(ws + 0 * tsz);
    __bf16* QS  = (__bf16*)(ws + 1 * tsz);
    __bf16* KC  = (__bf16*)(ws + 2 * tsz);
    __bf16* KS  = (__bf16*)(ws + 3 * tsz);
    __bf16* Vb  = (__bf16*)(ws + 4 * tsz);
    __bf16* ybf = (__bf16*)(ws + 5 * tsz);                          // 4 MB
    const size_t wsz = (size_t)DIM_ * DIM_ * sizeof(__bf16);        // 0.5 MB
    __bf16* Wqb = (__bf16*)(ws + 6 * tsz);
    __bf16* Wkb = (__bf16*)(ws + 6 * tsz + wsz);
    __bf16* Wvb = (__bf16*)(ws + 6 * tsz + 2 * wsz);

    // Pass 0: one-time f32 -> bf16 conversion of GEMM inputs
    const int yn8 = (N_ * L_ * DIM_) / 8;        // 262144
    const int wn8 = (DIM_ * DIM_) / 8;           // 32768
    fka_cvt_kernel<<<(yn8 + 255) / 256, 256, 0, stream>>>(y,  ybf, yn8);
    fka_cvt_kernel<<<(wn8 + 255) / 256, 256, 0, stream>>>(Wq, Wqb, wn8);
    fka_cvt_kernel<<<(wn8 + 255) / 256, 256, 0, stream>>>(Wk, Wkb, wn8);
    fka_cvt_kernel<<<(wn8 + 255) / 256, 256, 0, stream>>>(Wv, Wvb, wn8);

    // Pass 1: projections + rotation
    fka_proj_kernel<<<dim3((N_ * L_) / 64, DIM_ / 64, 3), 128, 0, stream>>>(
        ybf, Wqb, Wkb, Wvb, positions, coeffs, pw, pb, QC, QS, KC, KS, Vb);

    // Pass 2: causal normalized attention (128-row query blocks, 8 waves)
    fka_attn_kernel<<<dim3(L_ / 128, N_ * H_), 256, 0, stream>>>(
        QC, QS, KC, KS, Vb, (float*)d_out);
}
